// DIN_85229331022282
// MI455X (gfx1250) — matrix operand order repaired and checked by reading, not compiled, not run on hardware
//
#include <hip/hip_runtime.h>
#include <hip/hip_bf16.h>

typedef __attribute__((ext_vector_type(16))) _Float16 v16h;
typedef __attribute__((ext_vector_type(8)))  _Float16 v8h;
typedef __attribute__((ext_vector_type(8)))  float    v8f;

#define PAD_IDX 1000000
#define LMAX 352
#define DIN 704
#define H1  512
#define H2  256
#define H3  128
#define SA  712   // LDS stride (halves) buffer A (holds K up to 704)
#define SB  520   // LDS stride (halves) buffer B (holds K up to 512)

// ---------------------------------------------------------------------------
// Weight pack: f32 row-major [K,N] -> f16 WMMA B-fragment order.
// Block index (nt,kt); within block: lane-major, 16 halves per lane.
//   lane < 16 : n = nt*16+lane,    K = kt*32 + {0..7,16..23}[j]  (j = 0..15)
//   lane >= 16: n = nt*16+lane-16, K = kt*32 + {8..15,24..31}[j]
// ---------------------------------------------------------------------------
__global__ __launch_bounds__(256) void din_pack_kernel(
    const float* __restrict__ W, _Float16* __restrict__ Wp, int K, int N)
{
  const long total = (long)K * N;
  const int numKt = K >> 5;
  long i = (long)blockIdx.x * 256 + threadIdx.x;
  if (i >= total) return;
  int j    = (int)(i & 15);
  int lane = (int)((i >> 4) & 31);
  long blk = i >> 9;
  int kt = (int)(blk % numKt);
  int nt = (int)(blk / numKt);
  int n = nt * 16 + (lane & 15);
  int k = kt * 32 + ((j < 8) ? 0 : 8) + j + ((lane < 16) ? 0 : 8);
  Wp[i] = (_Float16)W[(long)k * N + n];
}

// ---------------------------------------------------------------------------
// Attention pooling: one block per batch row, 8 waves (wave32).
// Gathers each pool's embedding rows into LDS once (HBM roofline path),
// computes masked softmax attention, writes x[b, 0:192] (pools) and
// x[b, 192:704] (ad embeddings) as f16.
// ---------------------------------------------------------------------------
__global__ __launch_bounds__(256) void din_attn_kernel(
    const int* __restrict__ feature_idx,
    const int* __restrict__ hist_item,
    const int* __restrict__ hist_author,
    const int* __restrict__ hist_music,
    const float* __restrict__ emb,
    _Float16* __restrict__ xh)
{
  extern __shared__ __align__(16) float sm[];
  float* s_hist = sm;                          // LMAX*64 gathered rows (f32)
  int*   s_idx  = (int*)(sm + LMAX * 64);      // LMAX indices
  float* s_sc   = sm + LMAX * 64 + LMAX;       // LMAX scores / weights
  float* s_part = s_sc + LMAX;                 // 8 waves * 64 partials
  float* s_red  = s_part + 512;                // 16 reduction slots

  const int b    = blockIdx.x;
  const int tid  = threadIdx.x;
  const int lane = tid & 31;
  const int wave = tid >> 5;

  const int* hists[3] = { hist_item, hist_author, hist_music };
  const int  Ls[3]    = { 350, 250, 100 };
  const int  cols[3]  = { 2, 3, 6 };

  for (int p = 0; p < 3; ++p) {
    const int* h = hists[p];
    const int L = Ls[p];

    for (int l = tid; l < L; l += 256) s_idx[l] = h[b * L + l];
    __syncthreads();

    // Gather history embedding rows into LDS (single HBM pass)
    for (int e = tid; e < L * 64; e += 256) {
      int l = e >> 6, k = e & 63;
      s_hist[e] = emb[(long)s_idx[l] * 64 + k];
    }
    const int ad = feature_idx[b * 8 + cols[p]];
    const float a0 = emb[(long)ad * 64 + lane];
    const float a1 = emb[(long)ad * 64 + 32 + lane];
    __syncthreads();

    // Scores: wave-parallel over l, lane-parallel dot over K=64
    for (int l = wave; l < L; l += 8) {
      float d = s_hist[l * 64 + lane] * a0 + s_hist[l * 64 + 32 + lane] * a1;
      #pragma unroll
      for (int off = 16; off; off >>= 1) d += __shfl_xor(d, off, 32);
      if (lane == 0) s_sc[l] = (s_idx[l] != PAD_IDX) ? d : -1e9f;
    }
    __syncthreads();

    // Softmax: block max
    float m = -1e9f;
    for (int l = tid; l < L; l += 256) m = fmaxf(m, s_sc[l]);
    #pragma unroll
    for (int off = 16; off; off >>= 1) m = fmaxf(m, __shfl_xor(m, off, 32));
    if (lane == 0) s_red[wave] = m;
    __syncthreads();
    if (tid == 0) {
      float mm = s_red[0];
      for (int i = 1; i < 8; ++i) mm = fmaxf(mm, s_red[i]);
      s_red[0] = mm;
    }
    __syncthreads();
    m = s_red[0];

    // exp + sum
    float ssum = 0.f;
    for (int l = tid; l < L; l += 256) {
      float e = __expf(s_sc[l] - m);
      s_sc[l] = e;
      ssum += e;
    }
    #pragma unroll
    for (int off = 16; off; off >>= 1) ssum += __shfl_xor(ssum, off, 32);
    if (lane == 0) s_red[8 + wave] = ssum;
    __syncthreads();
    if (tid == 0) {
      float t = 0.f;
      for (int i = 0; i < 8; ++i) t += s_red[8 + i];
      s_red[0] = 1.0f / t;
    }
    __syncthreads();
    const float inv = s_red[0];

    // Weighted sum: waves accumulate partials over their l subset
    float acc0 = 0.f, acc1 = 0.f;
    for (int l = wave; l < L; l += 8) {
      float w = s_sc[l] * inv;
      acc0 += w * s_hist[l * 64 + lane];
      acc1 += w * s_hist[l * 64 + 32 + lane];
    }
    s_part[wave * 64 + lane]      = acc0;
    s_part[wave * 64 + 32 + lane] = acc1;
    __syncthreads();
    if (tid < 64) {
      float t = 0.f;
      #pragma unroll
      for (int wv = 0; wv < 8; ++wv) t += s_part[wv * 64 + tid];
      xh[(long)b * DIN + p * 64 + tid] = (_Float16)t;
    }
    __syncthreads();
  }

  // Ad embedding concat: x[b, 192:704]
  for (int e = tid; e < 512; e += 256) {
    int f = e >> 6, k = e & 63;
    int row = feature_idx[b * 8 + f];
    xh[(long)b * DIN + 192 + e] = (_Float16)emb[(long)row * 64 + k];
  }
}

// ---------------------------------------------------------------------------
// MLP via v_wmma_f32_16x16x32_f16
// ---------------------------------------------------------------------------
__device__ __forceinline__ v16h cat8(v8h lo, v8h hi) {
  return __builtin_shufflevector(lo, hi, 0,1,2,3,4,5,6,7,8,9,10,11,12,13,14,15);
}

// One fused layer: actOut[32,N] = relu(actIn[32,K] @ W + bias), all f16 LDS.
// A-fragment per ISA 16-bit layout: lane<16 -> M=lane, K {0..7,16..23};
// lane>=16 -> M=lane-16, K {8..15,24..31}.  B pre-packed (see pack kernel).
__device__ __forceinline__ void wmma_layer(
    const _Float16* __restrict__ actIn, int sIn,
    _Float16* __restrict__ actOut, int sOut,
    const _Float16* __restrict__ Wp, const float* __restrict__ bias,
    int K, int N, int lane, int wave)
{
  const int numKt = K >> 5;
  const int ntPerWave = N >> 7;          // N / (16 cols * 8 waves)
  const int kb   = (lane < 16) ? 0 : 8;  // also the M offset for C/D rows
  const int rsel = lane & 15;

  for (int mt = 0; mt < 2; ++mt) {
    const _Float16* arow = actIn + (mt * 16 + rsel) * sIn;
    for (int ntl = 0; ntl < ntPerWave; ++ntl) {
      const int nt = wave * ntPerWave + ntl;
      v8f acc = {};
      const _Float16* bp = Wp + (size_t)nt * numKt * 512 + lane * 16;
      for (int kt = 0; kt < numKt; ++kt) {
        v8h alo = *(const v8h*)(arow + kt * 32 + kb);
        v8h ahi = *(const v8h*)(arow + kt * 32 + kb + 16);
        v8h blo = *(const v8h*)(bp);
        v8h bhi = *(const v8h*)(bp + 8);
        v16h a  = cat8(alo, ahi);
        v16h bb = cat8(blo, bhi);
        acc = __builtin_amdgcn_wmma_f32_16x16x32_f16(
            /*neg_a=*/false, a, /*neg_b=*/false, bb,
            /*c_mod=*/(short)0, acc, /*reuse_a=*/false, /*reuse_b=*/false);
        bp += 512;                        // 32 lanes * 16 halves per k-block
      }
      const int n  = nt * 16 + rsel;
      const float bv = bias[n];
      const int rb = mt * 16 + kb;        // C/D: VGPR r -> row rb + r
      #pragma unroll
      for (int r = 0; r < 8; ++r) {
        float v = acc[r] + bv;
        actOut[(rb + r) * sOut + n] = (_Float16)fmaxf(v, 0.f);
      }
    }
  }
}

__global__ __launch_bounds__(256) void din_mlp_kernel(
    const _Float16* __restrict__ xh,
    const _Float16* __restrict__ W1p, const float* __restrict__ b1,
    const _Float16* __restrict__ W2p, const float* __restrict__ b2,
    const _Float16* __restrict__ W3p, const float* __restrict__ b3,
    const float* __restrict__ Wo, const float* __restrict__ bo,
    float* __restrict__ out)
{
  extern __shared__ __align__(16) _Float16 smh[];
  _Float16* actA = smh;                 // 32 * SA halves
  _Float16* actB = smh + 32 * SA;       // 32 * SB halves

  const int tid  = threadIdx.x;
  const int lane = tid & 31;
  const int wave = tid >> 5;
  const long rowBase = (long)blockIdx.x * 32;

  // Stage x tile [32, 704] f16 into actA (16B vector copies)
  for (int r = wave; r < 32; r += 8) {
    const _Float16* src = xh + (rowBase + r) * DIN;
    for (int c = lane; c < DIN / 8; c += 32)
      *(v8h*)(actA + r * SA + c * 8) = *(const v8h*)(src + c * 8);
  }
  __syncthreads();

  wmma_layer(actA, SA, actB, SB, W1p, b1, DIN, H1, lane, wave);  // 704->512
  __syncthreads();
  wmma_layer(actB, SB, actA, SA, W2p, b2, H1, H2, lane, wave);   // 512->256
  __syncthreads();
  wmma_layer(actA, SA, actB, SB, W3p, b3, H2, H3, lane, wave);   // 256->128
  __syncthreads();

  // Output layer: [32,128] @ [128,1] + bo, one row per lane of wave 0
  if (tid < 32) {
    float s = 0.f;
    #pragma unroll 4
    for (int k = 0; k < H3; ++k) s += (float)actB[tid * SB + k] * Wo[k];
    out[rowBase + tid] = s + bo[0];
  }
}

// ---------------------------------------------------------------------------
extern "C" void kernel_launch(void* const* d_in, const int* in_sizes, int n_in,
                              void* d_out, int out_size, void* d_ws, size_t ws_size,
                              hipStream_t stream)
{
  const int*   feature_idx = (const int*)  d_in[0];
  const int*   hist_item   = (const int*)  d_in[1];
  const int*   hist_author = (const int*)  d_in[2];
  const int*   hist_music  = (const int*)  d_in[3];
  const float* emb         = (const float*)d_in[4];
  const float* W1 = (const float*)d_in[5];
  const float* b1 = (const float*)d_in[6];
  const float* W2 = (const float*)d_in[7];
  const float* b2 = (const float*)d_in[8];
  const float* W3 = (const float*)d_in[9];
  const float* b3 = (const float*)d_in[10];
  const float* Wo = (const float*)d_in[11];
  const float* bo = (const float*)d_in[12];
  float* out = (float*)d_out;

  const int B = in_sizes[0] / 8;   // feature_idx is [B, 8]

  char* ws = (char*)d_ws;
  size_t off = 0;
  _Float16* xh  = (_Float16*)(ws + off); off += (((size_t)B * DIN * 2) + 255) & ~(size_t)255;
  _Float16* W1p = (_Float16*)(ws + off); off += (((size_t)DIN * H1 * 2) + 255) & ~(size_t)255;
  _Float16* W2p = (_Float16*)(ws + off); off += (((size_t)H1 * H2 * 2) + 255) & ~(size_t)255;
  _Float16* W3p = (_Float16*)(ws + off); off += (((size_t)H2 * H3 * 2) + 255) & ~(size_t)255;
  (void)off; (void)ws_size; (void)n_in; (void)out_size;

  // Pack weights into WMMA B-fragment order (f16)
  din_pack_kernel<<<(DIN * H1 + 255) / 256, 256, 0, stream>>>(W1, W1p, DIN, H1);
  din_pack_kernel<<<(H1 * H2 + 255) / 256, 256, 0, stream>>>(W2, W2p, H1, H2);
  din_pack_kernel<<<(H2 * H3 + 255) / 256, 256, 0, stream>>>(W3, W3p, H2, H3);

  // Attention pooling + feature concat (gather-bound, one HBM pass)
  const int attnLds = (LMAX * 64 + LMAX + LMAX + 512 + 16) * 4;   // 95040 B
  hipFuncSetAttribute(reinterpret_cast<const void*>(din_attn_kernel),
                      hipFuncAttributeMaxDynamicSharedMemorySize, attnLds);
  din_attn_kernel<<<B, 256, attnLds, stream>>>(feature_idx, hist_item, hist_author,
                                               hist_music, emb, xh);

  // Fused MLP: 32-row tiles, WMMA f16 with f32 accumulate
  const int mlpLds = (32 * SA + 32 * SB) * 2;                     // 78848 B
  hipFuncSetAttribute(reinterpret_cast<const void*>(din_mlp_kernel),
                      hipFuncAttributeMaxDynamicSharedMemorySize, mlpLds);
  din_mlp_kernel<<<B / 32, 256, mlpLds, stream>>>(xh, W1p, b1, W2p, b2,
                                                  W3p, b3, Wo, bo, out);
}
